// GATNet_84224308675025
// MI455X (gfx1250) — compile-verified, matrix operand order checked by reading, not run on hardware
//
#include <hip/hip_runtime.h>
#include <hip/hip_bf16.h>
#include <math.h>

#define NGRAPHS 128

// ---------------- WMMA types ----------------
typedef __attribute__((ext_vector_type(16))) __bf16 v16bf;
typedef __attribute__((ext_vector_type(8)))  __bf16 v8bf;
typedef __attribute__((ext_vector_type(8)))  float  v8f;

__device__ __forceinline__ __bf16 f32_to_bf16(float f) {
  unsigned int u = __float_as_uint(f);
  u = u + 0x7FFFu + ((u >> 16) & 1u);          // round-to-nearest-even
  unsigned short hs = (unsigned short)(u >> 16);
  __bf16 r;
  __builtin_memcpy(&r, &hs, sizeof(r));
  return r;
}

// ---------------- pack kernels (pad + convert, one pass) ----------------
// A: f32 [M,K] -> bf16 [Mp,Kp] row-major, zero padded
__global__ void pack_a_kernel(const float* __restrict__ src, __bf16* __restrict__ dst,
                              int M, int K, int Kp, size_t total) {
  size_t i = (size_t)blockIdx.x * blockDim.x + threadIdx.x;
  if (i >= total) return;
  int r = (int)(i / Kp), c = (int)(i % Kp);
  float v = (r < M && c < K) ? src[(size_t)r * K + c] : 0.0f;
  dst[i] = f32_to_bf16(v);
}
// W: f32 [K,N] -> bf16 [Np,Kp] TRANSPOSED (row n holds k-contiguous data), zero padded
__global__ void pack_wt_kernel(const float* __restrict__ W, __bf16* __restrict__ dst,
                               int K, int N, int Kp, size_t total) {
  size_t i = (size_t)blockIdx.x * blockDim.x + threadIdx.x;
  if (i >= total) return;
  int n = (int)(i / Kp), k = (int)(i % Kp);
  float v = (k < K && n < N) ? W[(size_t)k * N + n] : 0.0f;
  dst[i] = f32_to_bf16(v);
}

// ---------------- WMMA GEMM on packed bf16 ----------------
// C[M,N] = act(A @ W + bias (+bias2) (+C)),  A packed [Mp,Kp], Wt packed [Np,Kp].
// Macro-tile 128x64, BK=32, 256 thr = 8 waves. Wave w owns rows [16w,16w+16),
// sweeps 64 cols -> 4 v_wmma / K-step with A-fragment reuse.
// ISA fragment pattern: lane (m=l&15, kh=(l>=16)*8) needs k in
// [kh,kh+8) u [16+kh,16+kh+8) -> two 16B chunks. A read direct from global
// (each element used once per block); B staged in LDS (shared by 8 waves).
#define BM 128
#define BN 64
#define LDT 40   // padded LDS row stride in bf16 (80B) -> conflict-free b128 reads

__global__ __launch_bounds__(256)
void wmma_gemm_pk(const __bf16* __restrict__ Ap,
                  const __bf16* __restrict__ Bp,
                  const float* __restrict__ bias,
                  const float* __restrict__ bias2,
                  float* __restrict__ C,
                  int M, int N, int Kp, int flags) {
  __shared__ __align__(16) __bf16 Bs[BN * LDT];

  const int tid  = threadIdx.x;
  const int lane = tid & 31;
  const int wv   = tid >> 5;                  // wave 0..7
  const int m0   = blockIdx.y * BM;
  const int n0   = blockIdx.x * BN;
  const int lm   = lane & 15;
  const int kh   = (lane & 16) ? 8 : 0;

  // B staging assignment: thread t copies 8 bf16 of row sn
  const int sn = tid >> 2;
  const int sk = (tid & 3) * 8;
  const __bf16* arow = Ap + (size_t)(m0 + wv * 16 + lm) * Kp;
  const __bf16* brow = Bp + (size_t)(n0 + sn) * Kp + sk;

  v8f acc[4] = {v8f{}, v8f{}, v8f{}, v8f{}};

  for (int k0 = 0; k0 < Kp; k0 += 32) {
    // stage B tile (branchless b128 copy)
    *(v8bf*)&Bs[sn * LDT + sk] = *(const v8bf*)(brow + k0);
    // A fragment direct from global
    const v8bf alo = *(const v8bf*)(arow + k0 + kh);
    const v8bf ahi = *(const v8bf*)(arow + k0 + 16 + kh);
    const v16bf af = __builtin_shufflevector(alo, ahi,
        0, 1, 2, 3, 4, 5, 6, 7, 8, 9, 10, 11, 12, 13, 14, 15);
    __syncthreads();
#pragma unroll
    for (int bn = 0; bn < 4; ++bn) {
      const __bf16* br = &Bs[(bn * 16 + lm) * LDT];
      const v8bf blo = *(const v8bf*)(br + kh);
      const v8bf bhi = *(const v8bf*)(br + 16 + kh);
      const v16bf bfr = __builtin_shufflevector(blo, bhi,
          0, 1, 2, 3, 4, 5, 6, 7, 8, 9, 10, 11, 12, 13, 14, 15);
      acc[bn] = __builtin_amdgcn_wmma_f32_16x16x32_bf16(false, af, false, bfr,
                                                        (short)0, acc[bn], false, false);
    }
    __syncthreads();
  }

  // epilogue (guards only here, post-WMMA)
#pragma unroll
  for (int bn = 0; bn < 4; ++bn) {
    const int col = n0 + bn * 16 + lm;
    if (col < N) {
      const float bv  = bias  ? bias[col]  : 0.0f;
      const float bv2 = bias2 ? bias2[col] : 0.0f;
#pragma unroll
      for (int r = 0; r < 8; ++r) {
        const int row = m0 + wv * 16 + r + ((lane & 16) ? 8 : 0);
        if (row < M) {
          float v = acc[bn][r] + bv + bv2;
          if (flags & 1) v += C[(size_t)row * N + col];
          if (flags & 2) v  = fmaxf(v, 0.0f);
          if (flags & 4) v  = tanhf(v);
          C[(size_t)row * N + col] = v;
        }
      }
    }
  }
}

// ---------------- elementwise helpers ----------------
__global__ void fill_kernel(float* __restrict__ p, float v, size_t n) {
  size_t i = (size_t)blockIdx.x * blockDim.x + threadIdx.x;
  if (i < n) p[i] = v;
}
__global__ void relu_kernel(float* __restrict__ p, size_t n) {
  size_t i = (size_t)blockIdx.x * blockDim.x + threadIdx.x;
  if (i < n) p[i] = fmaxf(p[i], 0.0f);
}
__global__ void gate_kernel(const float* __restrict__ z, const float* __restrict__ y,
                            float* __restrict__ h, size_t n) {
  size_t i = (size_t)blockIdx.x * blockDim.x + threadIdx.x;
  if (i < n) {
    float s = 1.0f / (1.0f + expf(-z[i]));
    h[i] = s * y[i] + (1.0f - s) * h[i];
  }
}

__device__ __forceinline__ void atomicMaxF(float* addr, float val) {
  int* ai = (int*)addr;
  int old = __float_as_int(*addr);
  while (__int_as_float(old) < val) {
    int assumed = old;
    old = atomicCAS(ai, assumed, __float_as_int(val));
    if (old == assumed) break;
  }
}

// ---------------- edge attention (H=2 heads) ----------------
__global__ void edge_alpha_kernel(const float* __restrict__ q, const float* __restrict__ k,
                                  const int* __restrict__ src, const int* __restrict__ dst,
                                  int E, int Hd, int dout, float scale,
                                  float* __restrict__ alpha) {
  int e = blockIdx.x * blockDim.y + threadIdx.y;
  if (e >= E) return;
  const float* qd = q + (size_t)dst[e] * Hd;
  const float* ks = k + (size_t)src[e] * Hd;
  float a0 = 0.0f, a1 = 0.0f;
  for (int f = threadIdx.x; f < dout; f += 32) {
    a0 += qd[f] * ks[f];
    a1 += qd[dout + f] * ks[dout + f];
  }
  for (int off = 16; off > 0; off >>= 1) {
    a0 += __shfl_down(a0, off, 32);
    a1 += __shfl_down(a1, off, 32);
  }
  if (threadIdx.x == 0) {
    alpha[(size_t)e * 2 + 0] = a0 * scale;
    alpha[(size_t)e * 2 + 1] = a1 * scale;
  }
}
__global__ void edge_max_kernel(const float* __restrict__ alpha, const int* __restrict__ dst,
                                float* __restrict__ amax, int E) {
  int i = blockIdx.x * blockDim.x + threadIdx.x;
  if (i >= 2 * E) return;
  atomicMaxF(&amax[dst[i >> 1] * 2 + (i & 1)], alpha[i]);
}
__global__ void edge_expsum_kernel(float* __restrict__ alpha, const int* __restrict__ dst,
                                   const float* __restrict__ amax, float* __restrict__ denom,
                                   int E) {
  int i = blockIdx.x * blockDim.x + threadIdx.x;
  if (i >= 2 * E) return;
  int d2 = dst[i >> 1] * 2 + (i & 1);
  float ex = expf(alpha[i] - amax[d2]);
  alpha[i] = ex;
  atomicAdd(&denom[d2], ex);
}
__global__ void edge_agg_kernel(const float* __restrict__ ex, const float* __restrict__ denom,
                                const float* __restrict__ v,
                                const int* __restrict__ src, const int* __restrict__ dst,
                                float* __restrict__ out, int E, int Hd, int dout) {
  int e = blockIdx.x * blockDim.y + threadIdx.y;
  if (e >= E) return;
  int s = src[e], d = dst[e];
  float w0 = ex[(size_t)e * 2 + 0] / (denom[d * 2 + 0] + 1e-16f);
  float w1 = ex[(size_t)e * 2 + 1] / (denom[d * 2 + 1] + 1e-16f);
  const float* vs = v + (size_t)s * Hd;
  float* od = out + (size_t)d * dout;
  for (int f = threadIdx.x; f < dout; f += 32)
    atomicAdd(&od[f], 0.5f * (w0 * vs[f] + w1 * vs[dout + f]));
}

// ---------------- global max pool per graph ----------------
__global__ void pool_max_kernel(const float* __restrict__ h, const int* __restrict__ batch,
                                float* __restrict__ pool, int N, int d) {
  size_t i = (size_t)blockIdx.x * blockDim.x + threadIdx.x;
  if (i >= (size_t)N * d) return;
  int node = (int)(i / d), f = (int)(i % d);
  atomicMaxF(&pool[batch[node] * d + f], h[i]);
}

// ---------------- head: emb = concat(a_x*x, a_xq*xq) ----------------
__global__ void head_emb_kernel(const float* __restrict__ x, const float* __restrict__ xq,
                                const float* __restrict__ ax, const float* __restrict__ axq,
                                float* __restrict__ emb, int F) {
  int i = blockIdx.x * blockDim.x + threadIdx.x;
  if (i >= NGRAPHS * 128) return;
  int b = i / 128, f = i % 128;
  emb[(size_t)b * 2 * F + f]     = ax[b]  * x[(size_t)b * F + f];
  emb[(size_t)b * 2 * F + F + f] = axq[b] * xq[(size_t)b * F + f];
}

// ================= host side =================
struct Lin    { const float *W, *b; };
struct ConvP  { Lin q, k, v, skip; };
struct BranchP { ConvP conv[3]; Lin gate1, gate2; const float* gate_bias; Lin g1, g2; };
struct HeadP  { Lin ax1; const float* ax2W; Lin axq1; const float* axq2W; Lin fc1, fc2, out; };

struct Arena { char* base; size_t off; size_t cap; };
static float* an(Arena& a, size_t nfloats) {
  float* p = (float*)(a.base + a.off);
  a.off += ((nfloats * sizeof(float)) + 255) & ~(size_t)255;
  return p;
}
static __bf16* anb(Arena& a, size_t nbf) {
  __bf16* p = (__bf16*)(a.base + a.off);
  a.off += ((nbf * sizeof(__bf16)) + 255) & ~(size_t)255;
  return p;
}
static inline int rup(int x, int m) { return (x + m - 1) / m * m; }
static inline int eb(size_t n) { return (int)((n + 255) / 256); }

static void pack_a(const float* src, __bf16* dst, int M, int K, hipStream_t s) {
  const int Kp = rup(K, 32);
  const size_t total = (size_t)rup(M, BM) * Kp;
  pack_a_kernel<<<eb(total), 256, 0, s>>>(src, dst, M, K, Kp, total);
}
// packs W into wpk, then runs the packed GEMM
static void gemm_pk(const __bf16* Ap, const float* W, __bf16* wpk,
                    const float* b, const float* b2, float* C,
                    int M, int K, int N, int flags, hipStream_t s) {
  const int Kp = rup(K, 32), Np = rup(N, BN);
  const size_t wt = (size_t)Np * Kp;
  pack_wt_kernel<<<eb(wt), 256, 0, s>>>(W, wpk, K, N, Kp, wt);
  dim3 grd(Np / BN, rup(M, BM) / BM);
  wmma_gemm_pk<<<grd, 256, 0, s>>>(Ap, wpk, b, b2, C, M, N, Kp, flags);
}

static void parse_lin(void* const* din, int& i, Lin& l, bool bias) {
  l.W = (const float*)din[i++];
  l.b = bias ? (const float*)din[i++] : nullptr;
}
static void parse_branch(void* const* din, int& i, BranchP& p) {
  for (int c = 0; c < 3; ++c) {
    parse_lin(din, i, p.conv[c].q, true);
    parse_lin(din, i, p.conv[c].k, true);
    parse_lin(din, i, p.conv[c].v, true);
    parse_lin(din, i, p.conv[c].skip, true);
  }
  parse_lin(din, i, p.gate1, true);
  parse_lin(din, i, p.gate2, true);
  p.gate_bias = (const float*)din[i++];
  parse_lin(din, i, p.g1, true);
  parse_lin(din, i, p.g2, true);
}

static void run_branch(const float* x, const int* src, const int* dst, const int* batch,
                       int N, int E, int feat, const BranchP& P, float* xout,
                       __bf16* apk, __bf16* wpk,
                       Arena a /*by value: arena auto-reset*/, hipStream_t s) {
  const int d = feat * 4, Hd = 2 * d;
  float* h     = an(a, (size_t)N * d);
  float* y     = an(a, (size_t)N * d);
  float* z     = an(a, (size_t)N * d);
  float* ab0   = an(a, (size_t)N * Hd);     // q, later v
  float* ab1   = an(a, (size_t)N * Hd);     // k
  float* alpha = an(a, (size_t)E * 2);      // alpha, later ex
  float* amax  = an(a, (size_t)N * 2);
  float* denom = an(a, (size_t)N * 2);
  float* pool  = an(a, (size_t)NGRAPHS * d);
  float* g1o   = an(a, (size_t)NGRAPHS * 1024);
  const float scale = 1.0f / sqrtf((float)d);
  const dim3 wblk(32, 4);

  for (int L = 0; L < 3; ++L) {
    const float* in = (L == 0) ? x : h;
    const int din   = (L == 0) ? feat : d;
    const ConvP& c  = P.conv[L];
    pack_a(in, apk, N, din, s);                                   // shared by q/k/skip/v
    gemm_pk(apk, c.q.W, wpk, c.q.b, nullptr, ab0, N, din, Hd, 0, s);
    gemm_pk(apk, c.k.W, wpk, c.k.b, nullptr, ab1, N, din, Hd, 0, s);
    edge_alpha_kernel<<<dim3((E + 3) / 4), wblk, 0, s>>>(ab0, ab1, src, dst, E, Hd, d, scale, alpha);
    fill_kernel<<<eb((size_t)N * 2), 256, 0, s>>>(amax, -3.0e38f, (size_t)N * 2);
    fill_kernel<<<eb((size_t)N * 2), 256, 0, s>>>(denom, 0.0f, (size_t)N * 2);
    edge_max_kernel<<<eb((size_t)E * 2), 256, 0, s>>>(alpha, dst, amax, E);
    edge_expsum_kernel<<<eb((size_t)E * 2), 256, 0, s>>>(alpha, dst, amax, denom, E);
    gemm_pk(apk, c.skip.W, wpk, c.skip.b, nullptr, y, N, din, d, 0, s);   // y = skip
    gemm_pk(apk, c.v.W, wpk, c.v.b, nullptr, ab0, N, din, Hd, 0, s);      // v reuses q buffer
    edge_agg_kernel<<<dim3((E + 3) / 4), wblk, 0, s>>>(alpha, denom, ab0, src, dst, y, E, Hd, d);
    if (L < 2) relu_kernel<<<eb((size_t)N * d), 256, 0, s>>>(y, (size_t)N * d);
    if (L == 0) { float* t = h; h = y; y = t; }
    else {
      pack_a(y, apk, N, d, s);
      gemm_pk(apk, P.gate1.W, wpk, P.gate1.b, P.gate_bias, z, N, d, d, 0, s);
      pack_a(h, apk, N, d, s);
      gemm_pk(apk, P.gate2.W, wpk, P.gate2.b, nullptr, z, N, d, d, 1, s);  // accumulate
      gate_kernel<<<eb((size_t)N * d), 256, 0, s>>>(z, y, h, (size_t)N * d);
    }
  }
  fill_kernel<<<eb((size_t)NGRAPHS * d), 256, 0, s>>>(pool, -3.0e38f, (size_t)NGRAPHS * d);
  pool_max_kernel<<<eb((size_t)N * d), 256, 0, s>>>(h, batch, pool, N, d);
  pack_a(pool, apk, NGRAPHS, d, s);
  gemm_pk(apk, P.g1.W, wpk, P.g1.b, nullptr, g1o, NGRAPHS, d, 1024, 2 /*relu*/, s);
  pack_a(g1o, apk, NGRAPHS, 1024, s);
  gemm_pk(apk, P.g2.W, wpk, P.g2.b, nullptr, xout, NGRAPHS, 1024, 128, 0, s);
}

extern "C" void kernel_launch(void* const* d_in, const int* in_sizes, int n_in,
                              void* d_out, int out_size, void* d_ws, size_t ws_size,
                              hipStream_t stream) {
  (void)n_in; (void)out_size;
  const float* mol_x   = (const float*)d_in[0];
  const int*   mol_ei  = (const int*)d_in[1];
  const int*   mol_b   = (const int*)d_in[2];
  const float* cl_x    = (const float*)d_in[3];
  const int*   cl_ei   = (const int*)d_in[4];
  const int*   cl_b    = (const int*)d_in[5];
  const int N_mol = in_sizes[2], E_mol = in_sizes[1] / 2;
  const int f_mol = in_sizes[0] / N_mol;
  const int N_cl  = in_sizes[5], E_cl  = in_sizes[4] / 2;
  const int f_cl  = in_sizes[3] / N_cl;

  int i = 6;
  BranchP mp, cp;  HeadP hp;
  parse_branch(d_in, i, mp);
  parse_branch(d_in, i, cp);
  parse_lin(d_in, i, hp.ax1, true);  hp.ax2W  = (const float*)d_in[i++];
  parse_lin(d_in, i, hp.axq1, true); hp.axq2W = (const float*)d_in[i++];
  parse_lin(d_in, i, hp.fc1, true);
  parse_lin(d_in, i, hp.fc2, true);
  parse_lin(d_in, i, hp.out, true);

  Arena root{(char*)d_ws, 0, ws_size};
  float* xout  = an(root, (size_t)NGRAPHS * 128);   // persists across branches
  float* xqout = an(root, (size_t)NGRAPHS * 128);

  // shared pack scratch (sized for the largest A and W packs anywhere in the net)
  size_t apkN = 0, wpkN = 0;
  auto upd_a = [&](int M, int K) { size_t v = (size_t)rup(M, BM) * rup(K, 32); if (v > apkN) apkN = v; };
  auto upd_w = [&](int K, int N) { size_t v = (size_t)rup(N, BN) * rup(K, 32); if (v > wpkN) wpkN = v; };
  upd_a(N_mol, f_mol); upd_a(N_mol, 4 * f_mol);
  upd_a(N_cl,  f_cl);  upd_a(N_cl,  4 * f_cl);
  upd_a(NGRAPHS, 4 * f_mol); upd_a(NGRAPHS, 4 * f_cl);
  upd_a(NGRAPHS, 1024); upd_a(NGRAPHS, 2048); upd_a(NGRAPHS, 512);
  upd_a(NGRAPHS, 256);  upd_a(NGRAPHS, 128);  upd_a(NGRAPHS, 64);
  for (int br = 0; br < 2; ++br) {
    int f = br ? f_cl : f_mol, dd = 4 * f;
    upd_w(f, 2 * dd); upd_w(dd, 2 * dd); upd_w(f, dd); upd_w(dd, dd);
    upd_w(dd, 1024); upd_w(1024, 128);
  }
  upd_w(128, 64); upd_w(64, 1); upd_w(256, 2048); upd_w(2048, 512); upd_w(512, 1);
  __bf16* apk = anb(root, apkN);
  __bf16* wpk = anb(root, wpkN);

  run_branch(mol_x, mol_ei, mol_ei + E_mol, mol_b, N_mol, E_mol, f_mol, mp, xout,  apk, wpk, root, stream);
  run_branch(cl_x,  cl_ei,  cl_ei + E_cl,  cl_b,  N_cl,  E_cl,  f_cl,  cp, xqout, apk, wpk, root, stream);

  // --- head (arena reuse is safe: stream is in-order) ---
  Arena ha = root;
  float* t1  = an(ha, (size_t)NGRAPHS * 64);
  float* t1q = an(ha, (size_t)NGRAPHS * 64);
  float* ax  = an(ha, (size_t)NGRAPHS);
  float* axq = an(ha, (size_t)NGRAPHS);
  float* emb = an(ha, (size_t)NGRAPHS * 256);
  float* fa  = an(ha, (size_t)NGRAPHS * 2048);
  float* fb  = an(ha, (size_t)NGRAPHS * 512);

  pack_a(xout, apk, NGRAPHS, 128, stream);
  gemm_pk(apk, hp.ax1.W, wpk, hp.ax1.b, nullptr, t1, NGRAPHS, 128, 64, 4 /*tanh*/, stream);
  pack_a(t1, apk, NGRAPHS, 64, stream);
  gemm_pk(apk, hp.ax2W, wpk, nullptr, nullptr, ax, NGRAPHS, 64, 1, 0, stream);
  pack_a(xqout, apk, NGRAPHS, 128, stream);
  gemm_pk(apk, hp.axq1.W, wpk, hp.axq1.b, nullptr, t1q, NGRAPHS, 128, 64, 4, stream);
  pack_a(t1q, apk, NGRAPHS, 64, stream);
  gemm_pk(apk, hp.axq2W, wpk, nullptr, nullptr, axq, NGRAPHS, 64, 1, 0, stream);
  head_emb_kernel<<<eb((size_t)NGRAPHS * 128), 256, 0, stream>>>(xout, xqout, ax, axq, emb, 128);
  pack_a(emb, apk, NGRAPHS, 256, stream);
  gemm_pk(apk, hp.fc1.W, wpk, hp.fc1.b, nullptr, fa, NGRAPHS, 256, 2048, 2 /*relu*/, stream);
  pack_a(fa, apk, NGRAPHS, 2048, stream);
  gemm_pk(apk, hp.fc2.W, wpk, hp.fc2.b, nullptr, fb, NGRAPHS, 2048, 512, 2, stream);
  pack_a(fb, apk, NGRAPHS, 512, stream);
  gemm_pk(apk, hp.out.W, wpk, hp.out.b, nullptr, (float*)d_out, NGRAPHS, 512, 1, 0, stream);
}
